// GCNBaseline_34711925686446
// MI455X (gfx1250) — compile-verified, matrix operand order checked by reading, not run on hardware
//
#include <hip/hip_runtime.h>
#include <math.h>

// ---- problem constants (match reference) ----
#define NN   100000
#define EE   1600000
#define CC   64
#define KK   16
#define BB   16384
#define HH   128
#define EMB  32
#define LN_EPS 1e-5f

typedef __attribute__((ext_vector_type(16))) __bf16 bf16x16;
typedef __attribute__((ext_vector_type(8)))  float  floatx8;

__device__ __forceinline__ unsigned short f2bf(float f) {
  union { float f; unsigned u; } x; x.f = f;
  unsigned u = x.u;
  unsigned r = u + 0x7fffu + ((u >> 16) & 1u);   // round-to-nearest-even
  return (unsigned short)(r >> 16);
}

// ------------------------------------------------------------------
// Templated WMMA GEMM: Out[M, 16*NT] = A[M,K](f32->bf16) @ W[16*NT,K]^T
// mode 0: raw store of f32 accumulators
// mode 1: +bias, LayerNorm(gamma,beta), ReLU  (fused via LDS per-wave tile)
// One wave = 16 rows x (16*NT) cols. Block = 4 waves = 64 rows.
// Requires: M % 16 == 0, K % 32 == 0, 16-byte aligned A/W rows.
// ------------------------------------------------------------------
template <int NT>
__global__ __launch_bounds__(128)
void gemm_wmma_kernel(const float* __restrict__ A, int lda,
                      const unsigned short* __restrict__ W, int K,
                      int M,
                      const float* __restrict__ bias,
                      const float* __restrict__ gamma,
                      const float* __restrict__ beta,
                      float* __restrict__ Out, int ldo, int mode)
{
  __shared__ float lds[4 * 16 * (16 * NT + 4)];
  const int wave  = threadIdx.x >> 5;
  const int lane  = threadIdx.x & 31;
  const int mbase = (blockIdx.x * 4 + wave) * 16;
  if (mbase >= M) return;                 // whole-wave exit: EXEC stays all-ones

  const int rsel = lane & 15;             // row (A) / col (B) selector
  const int ksel = (lane >> 4) << 3;      // K sub-offset: 0 or 8

  floatx8 acc[NT] = {};

  const float* arow = A + (size_t)(mbase + rsel) * lda + ksel;
  for (int kk = 0; kk < K; kk += 32) {
    // A fragment: lanes 0-15 -> K {0..7,16..23}; lanes 16-31 -> K {8..15,24..31}
    union { bf16x16 v; unsigned short u[16]; } a;
    const float* ap = arow + kk;
    float4 f0 = *(const float4*)(ap);
    float4 f1 = *(const float4*)(ap + 4);
    float4 f2 = *(const float4*)(ap + 16);
    float4 f3 = *(const float4*)(ap + 20);
    a.u[0]=f2bf(f0.x); a.u[1]=f2bf(f0.y); a.u[2]=f2bf(f0.z); a.u[3]=f2bf(f0.w);
    a.u[4]=f2bf(f1.x); a.u[5]=f2bf(f1.y); a.u[6]=f2bf(f1.z); a.u[7]=f2bf(f1.w);
    a.u[8]=f2bf(f2.x); a.u[9]=f2bf(f2.y); a.u[10]=f2bf(f2.z); a.u[11]=f2bf(f2.w);
    a.u[12]=f2bf(f3.x); a.u[13]=f2bf(f3.y); a.u[14]=f2bf(f3.z); a.u[15]=f2bf(f3.w);

#pragma unroll
    for (int t = 0; t < NT; ++t) {
      // B = W^T : B-fragment lane pattern == A pattern applied to rows of W
      union { bf16x16 v; uint4 q[2]; } b;
      const unsigned short* wp = W + (size_t)(16 * t + rsel) * K + kk + ksel;
      b.q[0] = *(const uint4*)(wp);
      b.q[1] = *(const uint4*)(wp + 16);
      acc[t] = __builtin_amdgcn_wmma_f32_16x16x32_bf16(
          false, a.v, false, b.v, (short)0, acc[t], false, false);
    }
  }

  // C/D layout: lane L, vgpr j -> (row = j + 8*(L>>4), col = 16t + (L&15))
  const int rr = (lane >> 4) << 3;
  const int cc = lane & 15;

  if (mode == 0) {
#pragma unroll
    for (int t = 0; t < NT; ++t)
#pragma unroll
      for (int j = 0; j < 8; ++j)
        Out[(size_t)(mbase + rr + j) * ldo + 16 * t + cc] = acc[t][j];
    return;
  }

  // mode 1: bias + LayerNorm + ReLU, staged through per-wave LDS tile
  const int ncols = 16 * NT;
  const int ldst  = ncols + 4;
  float* tile = &lds[wave * 16 * ldst];
#pragma unroll
  for (int t = 0; t < NT; ++t) {
    float bv = bias[16 * t + cc];
#pragma unroll
    for (int j = 0; j < 8; ++j)
      tile[(rr + j) * ldst + 16 * t + cc] = acc[t][j] + bv;
  }
  // 2 lanes per row: lane L handles row L&15, half (L>>4)
  const int row  = lane & 15;
  const int half = lane >> 4;
  const int hc   = ncols >> 1;
  const float* lp = &tile[row * ldst + half * hc];
  float s = 0.f, s2 = 0.f;
#pragma unroll 4
  for (int c = 0; c < hc; ++c) { float v = lp[c]; s += v; s2 += v * v; }
  s  += __shfl_xor(s, 16, 32);
  s2 += __shfl_xor(s2, 16, 32);
  const float inv = 1.0f / (float)ncols;
  float mean = s * inv;
  float var  = s2 * inv - mean * mean;
  float rstd = rsqrtf(var + LN_EPS);
  float* orow = Out + (size_t)(mbase + row) * ldo + half * hc;
#pragma unroll 4
  for (int c = 0; c < hc; ++c) {
    int col = half * hc + c;
    float v = (lp[c] - mean) * rstd * gamma[col] + beta[col];
    orow[c] = fmaxf(v, 0.f);
  }
}

// ---- weight f32 -> bf16 (with column zero-padding) ----
__global__ __launch_bounds__(256)
void convert_pad_kernel(const float* __restrict__ src, unsigned short* __restrict__ dst,
                        int rows, int scols, int dcols)
{
  int tid = blockIdx.x * 256 + threadIdx.x;
  if (tid >= rows * dcols) return;
  int r = tid / dcols, c = tid - r * dcols;
  dst[tid] = (c < scols) ? f2bf(src[(size_t)r * scols + c]) : (unsigned short)0;
}

// ---- F = [census | sum(pop) | zero-pad to 96] ----
__global__ __launch_bounds__(256)
void build_f_kernel(const float* __restrict__ census, const float* __restrict__ pop,
                    float* __restrict__ F)
{
  int tid = blockIdx.x * 256 + threadIdx.x;
  if (tid >= NN * 96) return;
  int n = tid / 96, c = tid - n * 96;
  float v = 0.f;
  if (c < CC) v = census[(size_t)n * CC + c];
  else if (c == CC) {
    float sum = 0.f;
#pragma unroll
    for (int k = 0; k < KK; ++k) sum += pop[(size_t)n * KK + k];
    v = sum;
  }
  F[tid] = v;
}

// ---- degree / dinv ----
__global__ __launch_bounds__(256)
void deg_init_kernel(float* __restrict__ deg)
{
  int n = blockIdx.x * 256 + threadIdx.x;
  if (n < NN) deg[n] = 1.0f;             // self loop weight
}
__global__ __launch_bounds__(256)
void deg_edge_kernel(const int* __restrict__ dst, const float* __restrict__ ew,
                     float* __restrict__ deg)
{
  int e = blockIdx.x * 256 + threadIdx.x;
  if (e < EE) atomicAdd(&deg[dst[e]], ew[e]);
}
__global__ __launch_bounds__(256)
void dinv_kernel(const float* __restrict__ deg, float* __restrict__ dinv)
{
  int n = blockIdx.x * 256 + threadIdx.x;
  if (n < NN) dinv[n] = rsqrtf(deg[n]);   // deg >= 1 always
}

// ---- agg init with self loop: agg[n] = dinv[n]^2 * h[n] ----
__global__ __launch_bounds__(256)
void init_agg_kernel(const float* __restrict__ dinv, const float* __restrict__ h,
                     float* __restrict__ agg)
{
  int tid = blockIdx.x * 256 + threadIdx.x;   // NN*32 threads, float4 each
  if (tid >= NN * 32) return;
  int n = tid >> 5, c = (tid & 31) * 4;
  float dv = dinv[n];
  float nrm = dv * dv;
  float4 hv = *(const float4*)(h + (size_t)n * HH + c);
  float4 r; r.x = hv.x * nrm; r.y = hv.y * nrm; r.z = hv.z * nrm; r.w = hv.w * nrm;
  *(float4*)(agg + (size_t)n * HH + c) = r;
}

// ---- edge scatter: one wave per edge, 4 floats/lane, f32 atomics ----
__global__ __launch_bounds__(256)
void edge_scatter_kernel(const int* __restrict__ src, const int* __restrict__ dst,
                         const float* __restrict__ ew, const float* __restrict__ dinv,
                         const float* __restrict__ h, float* __restrict__ agg)
{
  int gw   = (blockIdx.x * 256 + threadIdx.x) >> 5;
  int lane = threadIdx.x & 31;
  if (gw >= EE) return;
  int s = src[gw], d = dst[gw];
  float norm = dinv[s] * ew[gw] * dinv[d];
  float4 hv = *(const float4*)(h + (size_t)s * HH + lane * 4);
  float* ap = agg + (size_t)d * HH + lane * 4;
  atomicAdd(ap + 0, hv.x * norm);
  atomicAdd(ap + 1, hv.y * norm);
  atomicAdd(ap + 2, hv.z * norm);
  atomicAdd(ap + 3, hv.w * norm);
}

// ---- post-agg: x = relu(ln(agg + bias)) (+ x residual). One wave per row. ----
__global__ __launch_bounds__(256)
void ln_res_kernel(const float* __restrict__ agg, const float* __restrict__ bias,
                   const float* __restrict__ g, const float* __restrict__ be,
                   float* __restrict__ x, int useRes)
{
  int row  = blockIdx.x * 8 + (threadIdx.x >> 5);
  int lane = threadIdx.x & 31;
  if (row >= NN) return;
  int c = lane * 4;
  float4 v  = *(const float4*)(agg + (size_t)row * HH + c);
  float4 bv = *(const float4*)(bias + c);
  v.x += bv.x; v.y += bv.y; v.z += bv.z; v.w += bv.w;
  float s  = v.x + v.y + v.z + v.w;
  float s2 = v.x * v.x + v.y * v.y + v.z * v.z + v.w * v.w;
#pragma unroll
  for (int m = 1; m < 32; m <<= 1) {
    s  += __shfl_xor(s, m, 32);
    s2 += __shfl_xor(s2, m, 32);
  }
  float mean = s * (1.f / HH);
  float var  = s2 * (1.f / HH) - mean * mean;
  float rstd = rsqrtf(var + LN_EPS);
  float4 gv  = *(const float4*)(g + c);
  float4 bev = *(const float4*)(be + c);
  float* xr  = x + (size_t)row * HH + c;
  float4 r;
  r.x = fmaxf((v.x - mean) * rstd * gv.x + bev.x, 0.f);
  r.y = fmaxf((v.y - mean) * rstd * gv.y + bev.y, 0.f);
  r.z = fmaxf((v.z - mean) * rstd * gv.z + bev.z, 0.f);
  r.w = fmaxf((v.w - mean) * rstd * gv.w + bev.w, 0.f);
  if (useRes) {
    float4 xo = *(const float4*)xr;
    r.x += xo.x; r.y += xo.y; r.z += xo.z; r.w += xo.w;
  }
  *(float4*)xr = r;
}

// ---- predictor gather: Hcat[b] = [x[node_idx[b]] (128) | eth_emb[eth[b]] (32)] ----
__global__ __launch_bounds__(256)
void gather_kernel(const float* __restrict__ x, const int* __restrict__ node_idx,
                   const int* __restrict__ eth, const float* __restrict__ emb,
                   float* __restrict__ Hcat)
{
  int tid = blockIdx.x * 256 + threadIdx.x;
  if (tid >= BB * 160) return;
  int b = tid / 160, c = tid - b * 160;
  float v;
  if (c < HH) v = x[(size_t)node_idx[b] * HH + c];
  else        v = emb[(size_t)eth[b] * EMB + (c - HH)];
  Hcat[tid] = v;
}

// ---- final dot(64) + softplus ----
__global__ __launch_bounds__(256)
void final_kernel(const float* __restrict__ P2, const float* __restrict__ w3,
                  const float* __restrict__ b3, float* __restrict__ out)
{
  int b = blockIdx.x * 256 + threadIdx.x;
  if (b >= BB) return;
  const float* p = P2 + (size_t)b * 64;
  float z = b3[0];
#pragma unroll 8
  for (int i = 0; i < 64; ++i) z += p[i] * w3[i];
  out[b] = fmaxf(z, 0.f) + log1pf(expf(-fabsf(z)));   // stable softplus
}

// ------------------------------------------------------------------
// workspace layout (bytes)
// ------------------------------------------------------------------
// bf16 weights (ushort element offsets within weight region)
#define WO_ENC1  0
#define WO_ENC2  (WO_ENC1 + 128*96)
#define WO_G0    (WO_ENC2 + 128*128)
#define WO_G1    (WO_G0   + 128*128)
#define WO_G2    (WO_G1   + 128*128)
#define WO_P1    (WO_G2   + 128*128)
#define WO_P2    (WO_P1   + 128*160)

#define OFF_XA   ((size_t)262144)
#define OFF_XB   (OFF_XA + (size_t)NN*HH*4)
#define OFF_XC   (OFF_XB + (size_t)NN*HH*4)
#define OFF_DEG  (OFF_XC + (size_t)NN*HH*4)
#define OFF_DINV (OFF_DEG + (size_t)NN*4)
#define WS_NEED  (OFF_DINV + (size_t)NN*4)

extern "C" void kernel_launch(void* const* d_in, const int* in_sizes, int n_in,
                              void* d_out, int out_size, void* d_ws, size_t ws_size,
                              hipStream_t stream)
{
  (void)in_sizes; (void)n_in; (void)out_size;
  if (ws_size < WS_NEED) return;

  const float* census  = (const float*)d_in[0];
  const float* pop     = (const float*)d_in[1];
  const int*   nodeIdx = (const int*)  d_in[2];
  const int*   eth     = (const int*)  d_in[3];
  const int*   eidx    = (const int*)  d_in[4];
  const int*   esrc    = eidx;
  const int*   edst    = eidx + EE;
  const float* ew      = (const float*)d_in[5];
  const float* encW1 = (const float*)d_in[6];
  const float* encb1 = (const float*)d_in[7];
  const float* encg1 = (const float*)d_in[8];
  const float* encbe1= (const float*)d_in[9];
  const float* encW2 = (const float*)d_in[10];
  const float* encb2 = (const float*)d_in[11];
  const float* encg2 = (const float*)d_in[12];
  const float* encbe2= (const float*)d_in[13];
  const float* gW[3]  = {(const float*)d_in[14], (const float*)d_in[18], (const float*)d_in[22]};
  const float* gb[3]  = {(const float*)d_in[15], (const float*)d_in[19], (const float*)d_in[23]};
  const float* gg[3]  = {(const float*)d_in[16], (const float*)d_in[20], (const float*)d_in[24]};
  const float* gbe[3] = {(const float*)d_in[17], (const float*)d_in[21], (const float*)d_in[25]};
  const float* embW  = (const float*)d_in[26];
  const float* prW1  = (const float*)d_in[27];
  const float* prb1  = (const float*)d_in[28];
  const float* prg1  = (const float*)d_in[29];
  const float* prbe1 = (const float*)d_in[30];
  const float* prW2  = (const float*)d_in[31];
  const float* prb2  = (const float*)d_in[32];
  const float* prg2  = (const float*)d_in[33];
  const float* prbe2 = (const float*)d_in[34];
  const float* prW3  = (const float*)d_in[35];
  const float* prb3  = (const float*)d_in[36];
  float* out = (float*)d_out;

  char* ws = (char*)d_ws;
  unsigned short* wq = (unsigned short*)ws;
  unsigned short* wEnc1 = wq + WO_ENC1;
  unsigned short* wEnc2 = wq + WO_ENC2;
  unsigned short* wG[3] = {wq + WO_G0, wq + WO_G1, wq + WO_G2};
  unsigned short* wP1   = wq + WO_P1;
  unsigned short* wP2   = wq + WO_P2;

  float* XA   = (float*)(ws + OFF_XA);
  float* XB   = (float*)(ws + OFF_XB);
  float* XC   = (float*)(ws + OFF_XC);
  float* deg  = (float*)(ws + OFF_DEG);
  float* dinv = (float*)(ws + OFF_DINV);
  // region overlays on XC (sequentially disjoint in time):
  float* F    = XC;                                   // [N,96] before encoder
  float* Hcat = XC;                                   // [B,160] after GCN
  float* P1   = (float*)((char*)XC + (size_t)BB*160*4);
  float* P2   = (float*)((char*)P1 + (size_t)BB*128*4);

  // 1) weight conversion to bf16 (padded)
  convert_pad_kernel<<<(128*96 +255)/256, 256, 0, stream>>>(encW1, wEnc1, 128, 65, 96);
  convert_pad_kernel<<<(128*128+255)/256, 256, 0, stream>>>(encW2, wEnc2, 128, 128, 128);
  for (int i = 0; i < 3; ++i)
    convert_pad_kernel<<<(128*128+255)/256, 256, 0, stream>>>(gW[i], wG[i], 128, 128, 128);
  convert_pad_kernel<<<(128*160+255)/256, 256, 0, stream>>>(prW1, wP1, 128, 160, 160);
  convert_pad_kernel<<<(64*128 +255)/256, 256, 0, stream>>>(prW2, wP2, 64, 128, 128);

  // 2) input features F = [census | other | pad]
  build_f_kernel<<<(NN*96+255)/256, 256, 0, stream>>>(census, pop, F);

  // 3) encoder: two fused GEMM+LN+ReLU layers
  gemm_wmma_kernel<8><<<(NN+63)/64, 128, 0, stream>>>(F, 96, wEnc1, 96, NN,
                                                      encb1, encg1, encbe1, XA, HH, 1);
  gemm_wmma_kernel<8><<<(NN+63)/64, 128, 0, stream>>>(XA, HH, wEnc2, HH, NN,
                                                      encb2, encg2, encbe2, XB, HH, 1);

  // 4) symmetric degree norm (edge weights are layer-invariant -> compute once)
  deg_init_kernel<<<(NN+255)/256, 256, 0, stream>>>(deg);
  deg_edge_kernel<<<(EE+255)/256, 256, 0, stream>>>(edst, ew, deg);
  dinv_kernel<<<(NN+255)/256, 256, 0, stream>>>(deg, dinv);

  // 5) GCN layers: x lives in XB; h -> XC; agg -> XA
  for (int i = 0; i < 3; ++i) {
    gemm_wmma_kernel<8><<<(NN+63)/64, 128, 0, stream>>>(XB, HH, wG[i], HH, NN,
                                                        nullptr, nullptr, nullptr, XC, HH, 0);
    init_agg_kernel<<<(NN*32+255)/256, 256, 0, stream>>>(dinv, XC, XA);
    edge_scatter_kernel<<<(EE*32)/256, 256, 0, stream>>>(esrc, edst, ew, dinv, XC, XA);
    ln_res_kernel<<<(NN+7)/8, 256, 0, stream>>>(XA, gb[i], gg[i], gbe[i], XB, i > 0);
  }

  // 6) predictor
  gather_kernel<<<(BB*160+255)/256, 256, 0, stream>>>(XB, nodeIdx, eth, embW, Hcat);
  gemm_wmma_kernel<8><<<(BB+63)/64, 128, 0, stream>>>(Hcat, 160, wP1, 160, BB,
                                                      prb1, prg1, prbe1, P1, HH, 1);
  gemm_wmma_kernel<4><<<(BB+63)/64, 128, 0, stream>>>(P1, HH, wP2, HH, BB,
                                                      prb2, prg2, prbe2, P2, 64, 1);
  final_kernel<<<(BB+255)/256, 256, 0, stream>>>(P2, prW3, prb3, out);
}
